// CosMultiHeadAttention_85478439125203
// MI455X (gfx1250) — matrix operand order repaired and checked by reading, not compiled, not run on hardware
//
#include <hip/hip_runtime.h>

// ---------------------------------------------------------------------------
// CosMultiHeadAttention for MI455X (gfx1250, wave32, WMMA).
//   out = (attn(relu(qWq^T+bq), relu(kWk^T+bk), vWv^T+bv)) Wo^T + bo
// Pipeline: 3x GEMM (bf16 wmma) -> per-(b,h) linear attention (wmma) -> GEMM.
// Workspace layout (bf16): Qh | Kh | Vh | Attn, each B*L*D elements.
// ---------------------------------------------------------------------------

typedef __attribute__((ext_vector_type(16))) __bf16 v16bf;
typedef __attribute__((ext_vector_type(8)))  __bf16 v8bf;
typedef __attribute__((ext_vector_type(8)))  float  v8f;
typedef __attribute__((ext_vector_type(4)))  float  v4f;

#define L_SEQ   4096
#define DMODEL  1024
#define NHEADS  16
#define HD      64
#define FDIM    128
#define PI_F    3.14159265358979323846f

// --- optional CDNA5 async global->LDS path (ASYNCcnt-tracked) ---------------
#if defined(__gfx1250__) && \
    __has_builtin(__builtin_amdgcn_global_load_async_to_lds_b128) && \
    __has_builtin(__builtin_amdgcn_s_wait_asynccnt)
#define HAVE_ASYNC_LDS 1
#else
#define HAVE_ASYNC_LDS 0
#endif

#if HAVE_ASYNC_LDS
// Builtin signature (from hipcc diagnostic): param0/1 are pointers to
// GCC-vector 'int __vector(4)' in addrspace(1) (global) / addrspace(3) (LDS),
// followed by two integer immediates (offset, cpol).
typedef int v4i_g __attribute__((vector_size(16)));
typedef __attribute__((address_space(1))) v4i_g* glb_b128_p;
typedef __attribute__((address_space(3))) v4i_g* lds_b128_p;

__device__ __forceinline__ void async_ld_b128(void* lds_dst, const void* gsrc) {
    __builtin_amdgcn_global_load_async_to_lds_b128(
        (glb_b128_p)(gsrc), (lds_b128_p)(lds_dst), /*imm offset*/0, /*cpol*/0);
}
__device__ __forceinline__ void wait_async0() {
    __builtin_amdgcn_s_wait_asynccnt(0);
}
#endif

// --- WMMA helpers -----------------------------------------------------------

__device__ __forceinline__ v8f wmma_bf16(v16bf a, v16bf b, v8f c) {
    // D = A(16x32 bf16) * B(32x16 bf16) + C(16x16 f32)
    return __builtin_amdgcn_wmma_f32_16x16x32_bf16(
        /*neg_a=*/false, a, /*neg_b=*/false, b,
        /*c_mod=*/(short)0, c, /*reuse_a=*/false, /*reuse_b=*/false);
}

// A fragment, 16-bit A-matrix 16x32 layout (ISA 7.12.2):
//   lanes 0-15 : e0..7 -> K=0..7,   e8..15 -> K=16..23
//   lanes16-31 : e0..7 -> K=8..15,  e8..15 -> K=24..31
__device__ __forceinline__ v16bf ld_frag_a(const __bf16* row, int half) {
    union { v16bf v; v8bf h[2]; } u;
    u.h[0] = *(const v8bf*)(row + 8 * half);
    u.h[1] = *(const v8bf*)(row + 16 + 8 * half);
    return u.v;
}

// B fragment, 16-bit B-matrix 32x16: lane = column, same K order as A:
// lanes 0-15 e0..7 K=0..7, e8..15 K=16..23; lanes 16-31 K=8..15, K=24..31.
__device__ __forceinline__ v16bf ld_frag_b(const __bf16* row, int half) {
    union { v16bf v; v8bf h[2]; } u;
    u.h[0] = *(const v8bf*)(row + 8 * half);
    u.h[1] = *(const v8bf*)(row + 16 + 8 * half);
    return u.v;
}

// --- GEMM: C[M,N] = act(A[M,K] * W[N,K]^T + bias[N]) ------------------------
// Block tile 128x256, K-tile 32, double-buffered LDS. 256 threads = 8 waves,
// wave -> 64x64 subtile (4x4 WMMA tiles, 16 wmma per K-step per wave).

#define BM 128
#define BN 256
#define BK 32
#define TS 40   // LDS row stride (bf16 elems): 32 + 8 pad; 80B = 5*16B aligned

__global__ __launch_bounds__(256)
void gemm_bias_act(const float* __restrict__ Af, const __bf16* __restrict__ Ab,
                   const float* __restrict__ W, const float* __restrict__ bias,
                   float* __restrict__ Cf, __bf16* __restrict__ Cb,
                   int M, int N, int K, int relu)
{
    __shared__ __align__(16) __bf16 As[2][BM][TS];
    __shared__ __align__(16) __bf16 Ws[2][BN][TS];

    const int tid  = threadIdx.x;
    const int lane = tid & 31;
    const int wv   = tid >> 5;
    const int half = lane >> 4;
    const int mn   = lane & 15;
    const int wm   = (wv & 1) * 64;   // wave M offset in tile
    const int wn   = (wv >> 1) * 64;  // wave N offset in tile
    const int rowA = blockIdx.x * BM;
    const int rowW = blockIdx.y * BN;

    const int arow  = tid >> 1;        // 0..127 : A tile row for this thread
    const int akoff = (tid & 1) * 16;  // 0 / 16 : A K-half
    const int wrow  = tid;             // 0..255 : W tile row for this thread

    v8f acc[4][4];
    const v8f vzero = {0.f, 0.f, 0.f, 0.f, 0.f, 0.f, 0.f, 0.f};
#pragma unroll
    for (int mt = 0; mt < 4; ++mt)
#pragma unroll
        for (int nt = 0; nt < 4; ++nt) acc[mt][nt] = vzero;

    const int nk = K / BK;

    // ---- prologue: stage tile 0 into buffer 0 ----
    {
        if (Ab) {
            const __bf16* src = Ab + (size_t)(rowA + arow) * K + akoff;
#if HAVE_ASYNC_LDS
            async_ld_b128(&As[0][arow][akoff], src);
            async_ld_b128(&As[0][arow][akoff + 8], src + 8);
#else
            *(v8bf*)&As[0][arow][akoff]     = *(const v8bf*)(src);
            *(v8bf*)&As[0][arow][akoff + 8] = *(const v8bf*)(src + 8);
#endif
        } else {
            const float* src = Af + (size_t)(rowA + arow) * K + akoff;
#pragma unroll
            for (int i = 0; i < 16; i += 4) {
                v4f x = *(const v4f*)(src + i);
                As[0][arow][akoff + i + 0] = (__bf16)x[0];
                As[0][arow][akoff + i + 1] = (__bf16)x[1];
                As[0][arow][akoff + i + 2] = (__bf16)x[2];
                As[0][arow][akoff + i + 3] = (__bf16)x[3];
            }
        }
        const float* src = W + (size_t)(rowW + wrow) * K;
#pragma unroll
        for (int i = 0; i < 32; i += 4) {
            v4f x = *(const v4f*)(src + i);
            Ws[0][wrow][i + 0] = (__bf16)x[0];
            Ws[0][wrow][i + 1] = (__bf16)x[1];
            Ws[0][wrow][i + 2] = (__bf16)x[2];
            Ws[0][wrow][i + 3] = (__bf16)x[3];
        }
    }
#if HAVE_ASYNC_LDS
    if (Ab) wait_async0();
#endif
    __syncthreads();

    for (int kt = 0; kt < nk; ++kt) {
        const int cur  = kt & 1;
        const int nxt  = cur ^ 1;
        const bool more = (kt + 1 < nk);

        // ---- issue next tile's global loads (regs or async-to-LDS) ----
        v4f aA[4];
        v8bf aB[2];
        v4f aW[8];
        (void)aB;
        if (more) {
            const int k0 = (kt + 1) * BK;
            if (Ab) {
                const __bf16* src = Ab + (size_t)(rowA + arow) * K + k0 + akoff;
#if HAVE_ASYNC_LDS
                async_ld_b128(&As[nxt][arow][akoff], src);
                async_ld_b128(&As[nxt][arow][akoff + 8], src + 8);
#else
                aB[0] = *(const v8bf*)(src);
                aB[1] = *(const v8bf*)(src + 8);
#endif
            } else {
                const float* src = Af + (size_t)(rowA + arow) * K + k0 + akoff;
#pragma unroll
                for (int i = 0; i < 4; ++i) aA[i] = *(const v4f*)(src + 4 * i);
            }
            const float* srcw = W + (size_t)(rowW + wrow) * K + k0;
#pragma unroll
            for (int i = 0; i < 8; ++i) aW[i] = *(const v4f*)(srcw + 4 * i);
        }

        // ---- compute on current buffer: 16 WMMA per wave ----
        v16bf af[4];
#pragma unroll
        for (int mt = 0; mt < 4; ++mt)
            af[mt] = ld_frag_a(&As[cur][wm + mt * 16 + mn][0], half);
#pragma unroll
        for (int nt = 0; nt < 4; ++nt) {
            v16bf bf_ = ld_frag_b(&Ws[cur][wn + nt * 16 + mn][0], half);
#pragma unroll
            for (int mt = 0; mt < 4; ++mt)
                acc[mt][nt] = wmma_bf16(af[mt], bf_, acc[mt][nt]);
        }

        // ---- commit next tile into the other buffer ----
        if (more) {
            if (Ab) {
#if !HAVE_ASYNC_LDS
                *(v8bf*)&As[nxt][arow][akoff]     = aB[0];
                *(v8bf*)&As[nxt][arow][akoff + 8] = aB[1];
#endif
            } else {
#pragma unroll
                for (int i = 0; i < 4; ++i) {
                    As[nxt][arow][akoff + 4 * i + 0] = (__bf16)aA[i][0];
                    As[nxt][arow][akoff + 4 * i + 1] = (__bf16)aA[i][1];
                    As[nxt][arow][akoff + 4 * i + 2] = (__bf16)aA[i][2];
                    As[nxt][arow][akoff + 4 * i + 3] = (__bf16)aA[i][3];
                }
            }
#pragma unroll
            for (int i = 0; i < 8; ++i) {
                Ws[nxt][wrow][4 * i + 0] = (__bf16)aW[i][0];
                Ws[nxt][wrow][4 * i + 1] = (__bf16)aW[i][1];
                Ws[nxt][wrow][4 * i + 2] = (__bf16)aW[i][2];
                Ws[nxt][wrow][4 * i + 3] = (__bf16)aW[i][3];
            }
        }
#if HAVE_ASYNC_LDS
        if (Ab && more) wait_async0();
#endif
        __syncthreads();
    }

    // ---- epilogue: bias (+relu), store ----
#pragma unroll
    for (int mt = 0; mt < 4; ++mt) {
#pragma unroll
        for (int nt = 0; nt < 4; ++nt) {
            const int gc = rowW + wn + nt * 16 + mn;
            const float bval = bias[gc];
#pragma unroll
            for (int r = 0; r < 8; ++r) {
                const int gr = rowA + wm + mt * 16 + r + 8 * half;
                float v = acc[mt][nt][r] + bval;
                if (relu) v = fmaxf(v, 0.0f);
                if (Cf) Cf[(size_t)gr * N + gc] = v;
                else    Cb[(size_t)gr * N + gc] = (__bf16)v;
            }
        }
    }
}

// --- Linear attention core, one workgroup per (b,h) -------------------------
// k_ = [kh*sin_l ; kh*cos_l] (L x 128), q_ likewise.
// kv = k_^T v (128x64) via WMMA; z_l = 1/max(q_l . ksum, eps);
// attn_l = z_l * (q_l . kv) (64) via WMMA.

#define TL   64   // L-tile for pass A
#define KS_S 72   // ks/vs/qs LDS row stride (bf16): 144B = 9*16B aligned
#define KV_S 136  // kv_t LDS row stride (bf16):     272B = 17*16B aligned

__global__ __launch_bounds__(256)
void cos_attn_kernel(const __bf16* __restrict__ Qh,
                     const __bf16* __restrict__ Kh,
                     const __bf16* __restrict__ Vh,
                     __bf16* __restrict__ Attn)
{
    __shared__ __align__(16) __bf16 ks_t[FDIM][KS_S];   // k_^T tile: [feat][l]
    __shared__ __align__(16) __bf16 vs_t[HD][KS_S];     // v^T tile:  [d][l]
    __shared__ __align__(16) __bf16 kv_t[HD][KV_S];     // kv^T:      [d][feat]
    __shared__ float ksum_s[FDIM];
    __shared__ __align__(16) __bf16 qs[8][16][KS_S];    // per-wave q staging

    const int bh   = blockIdx.x;
    const int b    = bh >> 4;
    const int h    = bh & 15;
    const int tid  = threadIdx.x;
    const int lane = tid & 31;
    const int wv   = tid >> 5;
    const int half = lane >> 4;
    const int mn   = lane & 15;

    const size_t baseRow = (size_t)b * L_SEQ;
    const float  angScale = PI_F / (2.0f * (float)L_SEQ);

    // loader mapping for pass A
    const int l_in  = tid & 63;        // l within tile
    const int part  = tid >> 6;        // 0..3
    const int c0    = (part & 1) * 32; // head-col offset
    const int fbase = (part >> 1) * 64;// 0 = sin half, 64 = cos half

    v8f acck[4];
    const v8f vzero = {0.f, 0.f, 0.f, 0.f, 0.f, 0.f, 0.f, 0.f};
#pragma unroll
    for (int nt = 0; nt < 4; ++nt) acck[nt] = vzero;
    float ksum_r = 0.0f;

    // ======================= Pass A: kv and ksum ============================
    for (int lt = 0; lt < L_SEQ / TL; ++lt) {
        const int lg = lt * TL + l_in;
        const size_t rowoff = (baseRow + lg) * DMODEL + h * HD;
        const float ang  = angScale * (float)(lg + 1);
        const float trig = (fbase == 0) ? __sinf(ang) : __cosf(ang);
#pragma unroll 8
        for (int i = 0; i < 32; ++i) {
            float kvv = (float)Kh[rowoff + c0 + i];
            ks_t[fbase + c0 + i][l_in] = (__bf16)(kvv * trig);
        }
#pragma unroll 8
        for (int i = 0; i < 16; ++i)
            vs_t[part * 16 + i][l_in] = Vh[rowoff + part * 16 + i];
        __syncthreads();

        // wave wv owns feature rows [16*wv, 16*wv+16)
#pragma unroll
        for (int kt = 0; kt < 2; ++kt) {
            const int l0 = kt * 32;
            v16bf a = ld_frag_a(&ks_t[wv * 16 + mn][l0], half);
#pragma unroll
            for (int nt = 0; nt < 4; ++nt) {
                v16bf bmat = ld_frag_b(&vs_t[nt * 16 + mn][l0], half);
                acck[nt] = wmma_bf16(a, bmat, acck[nt]);
            }
        }
        // ksum over this tile (threads 0..127, one feature each)
        if (tid < FDIM) {
            float s = 0.0f;
#pragma unroll 8
            for (int l = 0; l < TL; ++l) s += (float)ks_t[tid][l];
            ksum_r += s;
        }
        __syncthreads();
    }

    // park kv (transposed [d][feat], bf16) + ksum in LDS
#pragma unroll
    for (int nt = 0; nt < 4; ++nt)
#pragma unroll
        for (int r = 0; r < 8; ++r)
            kv_t[nt * 16 + mn][wv * 16 + r + 8 * half] = (__bf16)acck[nt][r];
    if (tid < FDIM) ksum_s[tid] = ksum_r;
    __syncthreads();

    // ======================= Pass B: attn rows ==============================
    for (int mt = wv; mt < L_SEQ / 16; mt += 8) {
        const int lg = mt * 16 + mn;
        const size_t rowoff = (baseRow + lg) * DMODEL + h * HD;
        const int ch = half * 32;
        // stage this wave's 16x64 qh tile
#if HAVE_ASYNC_LDS
        async_ld_b128(&qs[wv][mn][ch],      Qh + rowoff + ch);
        async_ld_b128(&qs[wv][mn][ch + 16], Qh + rowoff + ch + 16);
        wait_async0();
#else
        *(v8bf*)&qs[wv][mn][ch]      = *(const v8bf*)(Qh + rowoff + ch);
        *(v8bf*)&qs[wv][mn][ch + 8]  = *(const v8bf*)(Qh + rowoff + ch + 8);
        *(v8bf*)&qs[wv][mn][ch + 16] = *(const v8bf*)(Qh + rowoff + ch + 16);
        *(v8bf*)&qs[wv][mn][ch + 24] = *(const v8bf*)(Qh + rowoff + ch + 24);
#endif

        const float ang  = angScale * (float)(lg + 1);
        const float sinl = __sinf(ang);
        const float cosl = __cosf(ang);

        // z for row mn (duplicated in both lane halves)
        float dlo = 0.0f, dhi = 0.0f;
#pragma unroll 8
        for (int c = 0; c < HD; ++c) {
            float qv = (float)qs[wv][mn][c];
            dlo += qv * ksum_s[c];
            dhi += qv * ksum_s[HD + c];
        }
        const float zr = 1.0f / fmaxf(sinl * dlo + cosl * dhi, 1e-6f);

        v8f acc[4];
#pragma unroll
        for (int nt = 0; nt < 4; ++nt) acc[nt] = vzero;

#pragma unroll
        for (int kt = 0; kt < 4; ++kt) {
            v16bf a;
#pragma unroll
            for (int e = 0; e < 16; ++e) {
                const int f = kt * 32 + (e < 8 ? 8 * half + e
                                               : 16 + 8 * half + (e - 8));
                const float qv = (float)qs[wv][mn][f & (HD - 1)];
                a[e] = (__bf16)(qv * ((f < HD) ? sinl : cosl));
            }
#pragma unroll
            for (int nt = 0; nt < 4; ++nt) {
                v16bf bmat = ld_frag_b(&kv_t[nt * 16 + mn][kt * 32], half);
                acc[nt] = wmma_bf16(a, bmat, acc[nt]);
            }
        }
        // scale by z and store (bf16) into Attn laid out as (B*L, D)
#pragma unroll
        for (int nt = 0; nt < 4; ++nt) {
#pragma unroll
            for (int r = 0; r < 8; ++r) {
                const int rm = r + 8 * half;
                const float zz = __shfl(zr, rm, 32);
                const size_t o = (baseRow + mt * 16 + rm) * DMODEL
                               + h * HD + nt * 16 + mn;
                Attn[o] = (__bf16)(acc[nt][r] * zz);
            }
        }
    }
}

// ---------------------------------------------------------------------------

extern "C" void kernel_launch(void* const* d_in, const int* in_sizes, int n_in,
                              void* d_out, int out_size, void* d_ws, size_t ws_size,
                              hipStream_t stream) {
    const float* q  = (const float*)d_in[0];
    const float* k  = (const float*)d_in[1];
    const float* v  = (const float*)d_in[2];
    const float* Wq = (const float*)d_in[3];
    const float* bq = (const float*)d_in[4];
    const float* Wk = (const float*)d_in[5];
    const float* bk = (const float*)d_in[6];
    const float* Wv = (const float*)d_in[7];
    const float* bv = (const float*)d_in[8];
    const float* Wo = (const float*)d_in[9];
    const float* bo = (const float*)d_in[10];
    float* out = (float*)d_out;

    const int B = in_sizes[0] / (L_SEQ * DMODEL);
    const int M = B * L_SEQ;
    const size_t nElem = (size_t)M * DMODEL;

    // workspace: 4 bf16 planes of B*L*D
    __bf16* Qh = (__bf16*)d_ws;
    __bf16* Kh = (__bf16*)((char*)d_ws + 1 * nElem * sizeof(__bf16));
    __bf16* Vh = (__bf16*)((char*)d_ws + 2 * nElem * sizeof(__bf16));
    __bf16* At = (__bf16*)((char*)d_ws + 3 * nElem * sizeof(__bf16));

    dim3 gg(M / BM, DMODEL / BN);
    gemm_bias_act<<<gg, 256, 0, stream>>>(q, nullptr, Wq, bq, nullptr, Qh,
                                          M, DMODEL, DMODEL, /*relu=*/1);
    gemm_bias_act<<<gg, 256, 0, stream>>>(k, nullptr, Wk, bk, nullptr, Kh,
                                          M, DMODEL, DMODEL, /*relu=*/1);
    gemm_bias_act<<<gg, 256, 0, stream>>>(v, nullptr, Wv, bv, nullptr, Vh,
                                          M, DMODEL, DMODEL, /*relu=*/0);
    cos_attn_kernel<<<B * NHEADS, 256, 0, stream>>>(Qh, Kh, Vh, At);
    gemm_bias_act<<<gg, 256, 0, stream>>>(nullptr, At, Wo, bo, out, nullptr,
                                          M, DMODEL, DMODEL, /*relu=*/0);
}
